// CRY_ENCODER_5076651344036
// MI455X (gfx1250) — matrix operand order repaired and checked by reading, not compiled, not run on hardware
//
#include <hip/hip_runtime.h>
#include <hip/hip_bf16.h>
#include <math.h>

// ---------------- problem constants ----------------
#define BB     16
#define TT     192
#define MROWS  (BB * TT)      // 3072
#define NE     512            // NEMBD
#define NH     8              // heads
#define HD     64             // head dim
#define EXTRA_ 363
#define CONVN  149            // NEMBD - EXTRA

typedef __attribute__((ext_vector_type(16))) _Float16 v16h;
typedef __attribute__((ext_vector_type(8)))  float    v8f;
typedef __attribute__((ext_vector_type(4)))  float    float4v;
typedef __attribute__((ext_vector_type(4)))  _Float16 half4;

// explicit global-address-space pointers so tile staging lowers to
// global_load_b128 (LOADcnt only) instead of flat_load_b128 (LOADcnt+DScnt)
typedef const __attribute__((address_space(1))) float    * gfp;
typedef const __attribute__((address_space(1))) float4v  * gf4p;

__device__ __forceinline__ float gelu_f(float x) {
  return 0.5f * x * (1.0f + erff(x * 0.70710678118654752440f));
}
__device__ __forceinline__ float softplus_f(float x) {
  return fmaxf(x, 0.0f) + log1pf(expf(-fabsf(x)));
}
__device__ __forceinline__ float sigmoid_f(float x) {
  return 1.0f / (1.0f + expf(-x));
}

// =====================================================================
// WMMA GEMM:  C[M,N] = epilogue(A[M,K] @ B[K,N] + bias)
// f32 inputs -> f16 tiles in LDS, f32 accumulation via
// v_wmma_f32_16x16x32_f16.  128 threads = 4 waves; block tile 64x128;
// each wave owns a 32x64 sub-tile (2x4 fragments); K-step 32.
// Per-thread global/LDS pointers hoisted out of the K loop, staged
// through address-space(1) so loads are pure global_load_b128.
// Template NG: ragged-N path (guards) -- only used by conv (N=149).
// =====================================================================
#define TM 64
#define TN 128
#define TK 32
#define AP 36    // A row stride (halves): 72 B  -> 8B-aligned rows, odd 8-byte stride
#define BP 132   // B row stride (halves): 264 B -> 8B-aligned rows, odd 8-byte stride

template <bool NG>
__global__ void __launch_bounds__(128)
gemm_wmma(const float* __restrict__ A, const float* __restrict__ Bw,
          const float* __restrict__ bias, const float* __restrict__ resid,
          const float* __restrict__ rowmask, float* __restrict__ C,
          int M, int N, int K, int lda, int ldb, int ldc, int act) {
  __shared__ alignas(16) _Float16 As[TM][AP];
  __shared__ alignas(16) _Float16 Bs[TK][BP];

  const int tid  = threadIdx.x;
  const int lane = tid & 31;
  const int wave = tid >> 5;
  const int wr   = (wave >> 1) * 32;     // wave row offset in block tile
  const int wc   = (wave & 1)  * 64;     // wave col offset
  const int m0   = blockIdx.y * TM;
  const int n0   = blockIdx.x * TN;
  const int ln15 = lane & 15;
  const int hi   = lane >> 4;            // 0 for lanes 0-15, 1 for 16-31

  // ---- loop-invariant staging pointers (global address space) ----
  gfp    apt[4];
  half4* asd[4];
#pragma unroll
  for (int i = 0; i < 4; ++i) {
    int idx = tid + i * 128;             // 0..511 : A tile 64x32
    int r = idx >> 3, c = (idx & 7) * 4;
    apt[i] = (gfp)(A + (size_t)(m0 + r) * lda + c);
    asd[i] = (half4*)&As[r][c];
  }
  gfp    bpt[8];
  half4* bsd[8];
#pragma unroll
  for (int i = 0; i < 8; ++i) {
    int idx = tid + i * 128;             // 0..1023 : B tile 32x128
    int r = idx >> 5, c = (idx & 31) * 4;
    bpt[i] = (gfp)(Bw + (size_t)r * ldb + n0 + c);
    bsd[i] = (half4*)&Bs[r][c];
  }
  const size_t bstep = (size_t)TK * ldb;
  // generic-pointer copies for prefetch only
  const float* apf = A + (size_t)(m0 + (tid >> 1)) * lda + TK;
  const float* bpf = Bw + (size_t)(tid >> 2) * ldb + n0;

  v8f acc[2][4];
  {
    v8f z = {};
#pragma unroll
    for (int mi = 0; mi < 2; ++mi)
#pragma unroll
      for (int ni = 0; ni < 4; ++ni) acc[mi][ni] = z;
  }

  for (int k0 = 0; k0 < K; k0 += TK) {
    // ---- stage A tile: 4 x global b128 loads, no guards ----
    float4v a4[4];
#pragma unroll
    for (int i = 0; i < 4; ++i) { a4[i] = *(gf4p)apt[i]; apt[i] += TK; }
    if (!NG) {
      // ---- stage B tile: 8 x global b128 loads ----
      float4v b4[8];
#pragma unroll
      for (int i = 0; i < 8; ++i) { b4[i] = *(gf4p)bpt[i]; bpt[i] += bstep; }
#pragma unroll
      for (int i = 0; i < 8; ++i) *bsd[i] = __builtin_convertvector(b4[i], half4);
    } else {
      for (int i = tid; i < TK * TN; i += 128) {
        int r = i >> 7, c = i & 127;
        int gn = n0 + c;
        float v = (gn < N) ? Bw[(size_t)(k0 + r) * ldb + gn] : 0.0f;
        Bs[r][c] = (_Float16)v;
      }
    }
#pragma unroll
    for (int i = 0; i < 4; ++i) *asd[i] = __builtin_convertvector(a4[i], half4);
    // prefetch next K tiles into cache while we compute this one
    if (k0 + TK < K) {
      __builtin_prefetch(apf, 0, 1);
      apf += TK;
      if (!NG) { __builtin_prefetch(bpf + (size_t)(k0 + TK) * ldb, 0, 1); }
    }
    __syncthreads();

    // ---- build fragments from LDS ----
    // A 16x32 f16 layout: lane = M%16; lanes 0-15: K = 0..7 & 16..23;
    // lanes 16-31: K = 8..15 & 24..31.
    v16h afrag[2], bfrag[4];
    const int lo = hi * 8;
#pragma unroll
    for (int mi = 0; mi < 2; ++mi) {
      int row = wr + mi * 16 + ln15;
#pragma unroll
      for (int h = 0; h < 8; ++h) {
        afrag[mi][h]     = As[row][lo + h];
        afrag[mi][h + 8] = As[row][16 + lo + h];
      }
    }
    // B 32x16 f16 layout: lane = N%16; lanes 0-15 hold K = 0..7 & 16..23,
    // lanes 16-31 hold K = 8..15 & 24..31.
    const int kb = hi * 8;
#pragma unroll
    for (int ni = 0; ni < 4; ++ni) {
      int col = wc + ni * 16 + ln15;
#pragma unroll
      for (int h = 0; h < 16; ++h) bfrag[ni][h] = Bs[(h < 8 ? kb + h : 16 + kb + h - 8)][col];
    }

#pragma unroll
    for (int mi = 0; mi < 2; ++mi)
#pragma unroll
      for (int ni = 0; ni < 4; ++ni)
        acc[mi][ni] = __builtin_amdgcn_wmma_f32_16x16x32_f16(
            /*neg_a=*/false, afrag[mi], /*neg_b=*/false, bfrag[ni],
            /*c_mod=*/(short)0, acc[mi][ni],
            /*reuse_a=*/false, /*reuse_b=*/false);
    __syncthreads();
  }

  // ---- epilogue ----
  // C/D layout: VGPR e: lanes 0-15 -> M = e, lanes 16-31 -> M = 8+e; N = lane%16
#pragma unroll
  for (int mi = 0; mi < 2; ++mi) {
#pragma unroll
    for (int ni = 0; ni < 4; ++ni) {
      int gn = n0 + wc + ni * 16 + ln15;
#pragma unroll
      for (int e = 0; e < 8; ++e) {
        int gm = m0 + wr + mi * 16 + hi * 8 + e;
        float val = acc[mi][ni][e];
        if (bias)     val += bias[gn];
        if (act == 1) val = gelu_f(val);
        if (resid)    val += resid[(size_t)gm * ldc + gn];
        if (rowmask)  val *= rowmask[gm];
        if (!NG || gn < N) C[(size_t)gm * ldc + gn] = val;
      }
    }
  }
}

// =====================================================================
// Embedding gather + validity mask
// =====================================================================
__global__ void __launch_bounds__(64)
embed_kernel(const int* __restrict__ nodes, const float* __restrict__ emb,
             float* __restrict__ h, float* __restrict__ valid) {
  int row = blockIdx.x;
  int c = threadIdx.x;
  int nd = nodes[row];
  h[(size_t)row * 64 + c] = emb[(size_t)nd * 64 + c];
  if (c == 0) valid[row] = (nd != 0) ? 1.0f : 0.0f;
}

// =====================================================================
// Fused CGCNN pair+gate+reduce.  One 64-thread block per (b,t); thread c
// owns channels (c, c+64) of the 128-wide pair vector; loops over s,
// never materializing the 300MB pair tensor.
// =====================================================================
__global__ void __launch_bounds__(64)
cgcnn_kernel(const float* __restrict__ hin, const float* __restrict__ HI,
             const float* __restrict__ HJ, const float* __restrict__ adj,
             const float* __restrict__ valid, const float* __restrict__ fcw,
             const float* __restrict__ bn1g, const float* __restrict__ bn1b,
             const float* __restrict__ bn2g, const float* __restrict__ bn2b,
             float* __restrict__ hout) {
  const int row = blockIdx.x;          // b*TT + t
  const int b = row / TT;
  const int c = threadIdx.x;           // 0..63

  float waf[12], wac[12];
#pragma unroll
  for (int k = 0; k < 12; ++k) {
    waf[k] = fcw[(size_t)(128 + k) * 128 + c];
    wac[k] = fcw[(size_t)(128 + k) * 128 + 64 + c];
  }
  const float inv = rsqrtf(1.0f + 1e-5f);          // bn: g / sqrt(1+eps)
  const float g1f = bn1g[c] * inv,      b1f = bn1b[c];
  const float g1c = bn1g[64 + c] * inv, b1c = bn1b[64 + c];
  const float hif = HI[(size_t)row * 128 + c];
  const float hic = HI[(size_t)row * 128 + 64 + c];
  const float vi  = valid[row];

  const float* adjrow = adj + (size_t)row * TT * 12;
  const float* hjb    = HJ + (size_t)(b * TT) * 128;
  const float* vb     = valid + b * TT;

  float accum = 0.0f;
  for (int s = 0; s < TT; ++s) {
    const float4v* ap = (const float4v*)(adjrow + (size_t)s * 12);
    float4v a0 = ap[0], a1 = ap[1], a2 = ap[2];
    float av[12] = {a0.x, a0.y, a0.z, a0.w, a1.x, a1.y, a1.z, a1.w,
                    a2.x, a2.y, a2.z, a2.w};
    float df = 0.0f, dc = 0.0f;
#pragma unroll
    for (int k = 0; k < 12; ++k) { df += av[k] * waf[k]; dc += av[k] * wac[k]; }
    float pf = hif + hjb[(size_t)s * 128 + c] + df;
    float pc = hic + hjb[(size_t)s * 128 + 64 + c] + dc;
    pf = pf * g1f + b1f;
    pc = pc * g1c + b1c;
    accum += sigmoid_f(pf) * softplus_f(pc) * (vi * vb[s]);
  }
  float agg = accum * (bn2g[c] * inv) + bn2b[c];
  hout[(size_t)row * 64 + c] = hin[(size_t)row * 64 + c] + softplus_f(agg);
}

__global__ void __launch_bounds__(64)
maskh_kernel(float* __restrict__ h, const float* __restrict__ valid) {
  int row = blockIdx.x;
  h[(size_t)row * 64 + threadIdx.x] *= valid[row];
}

__global__ void __launch_bounds__(128)
concat_kernel(const float* __restrict__ ne, float* __restrict__ t0) {
  int row = blockIdx.x;
  for (int c = threadIdx.x; c < EXTRA_; c += 128)
    t0[(size_t)row * NE + CONVN + c] = ne[(size_t)row * EXTRA_ + c];
}

// =====================================================================
// LayerNorm over last dim C (=512), one block per row
// =====================================================================
__global__ void __launch_bounds__(256)
ln_kernel(const float* __restrict__ x, const float* __restrict__ g,
          const float* __restrict__ b, float* __restrict__ y, int C) {
  __shared__ float red[256];
  const int row = blockIdx.x, tid = threadIdx.x;
  const float* xr = x + (size_t)row * C;

  float s = 0.0f;
  for (int c = tid; c < C; c += 256) s += xr[c];
  red[tid] = s; __syncthreads();
  for (int st = 128; st > 0; st >>= 1) {
    if (tid < st) red[tid] += red[tid + st];
    __syncthreads();
  }
  float mu = red[0] / (float)C;
  __syncthreads();

  float v = 0.0f;
  for (int c = tid; c < C; c += 256) { float d = xr[c] - mu; v += d * d; }
  red[tid] = v; __syncthreads();
  for (int st = 128; st > 0; st >>= 1) {
    if (tid < st) red[tid] += red[tid + st];
    __syncthreads();
  }
  float rstd = rsqrtf(red[0] / (float)C + 1e-5f);

  for (int c = tid; c < C; c += 256)
    y[(size_t)row * C + c] = (xr[c] - mu) * rstd * g[c] + b[c];
}

// =====================================================================
// Fused attention inner part for one (b, h, i) row:
//   scores -> mask -> softmax -> + aw * (gelu(adj@ap1)@ap2)[b,i,j,h] -> @V
// =====================================================================
__global__ void __launch_bounds__(192)
attn_kernel(const float* __restrict__ Q, const float* __restrict__ Kx,
            const float* __restrict__ V, const float* __restrict__ valid,
            const float* __restrict__ adj, const float* __restrict__ ap1,
            const float* __restrict__ ap2, const float* __restrict__ awp,
            float* __restrict__ Y) {
  const int i = blockIdx.x, h = blockIdx.y, b = blockIdx.z;
  const int j = threadIdx.x;
  __shared__ float qs[HD];
  __shared__ float atts[TT];
  __shared__ float red[256];
  __shared__ float ap1s[12 * 16];
  __shared__ float ap2s[16 * 8];

  const int rowi = b * TT + i;
  if (j < HD) qs[j] = Q[(size_t)rowi * NE + h * HD + j];
  if (j < 12 * 16) ap1s[j] = ap1[j];
  if (j < 16 * 8)  ap2s[j] = ap2[j];
  __syncthreads();

  // scores
  const float* kr = Kx + ((size_t)(b * TT + j) * NE + h * HD);
  float s = 0.0f;
#pragma unroll 8
  for (int d = 0; d < HD; ++d) s += qs[d] * kr[d];
  s *= 0.125f;                                    // 1/sqrt(64)
  float pm = valid[rowi] * valid[b * TT + j];
  if (pm == 0.0f) s = -1e9f;

  // softmax (192-wide reduction, pad to 256 with identities)
  red[j] = s;
  if (j < 64) red[192 + j] = -3.0e38f;
  __syncthreads();
  for (int st = 128; st > 0; st >>= 1) {
    if (j < st) red[j] = fmaxf(red[j], red[j + st]);
    __syncthreads();
  }
  float mx = red[0];
  __syncthreads();
  float e = expf(s - mx);
  red[j] = e;
  if (j < 64) red[192 + j] = 0.0f;
  __syncthreads();
  for (int st = 128; st > 0; st >>= 1) {
    if (j < st) red[j] += red[j + st];
    __syncthreads();
  }
  float att = e / red[0];

  // adjacency bias: gelu(adj[b,i,j,:] @ ap1) @ ap2[:, h]
  const float* av = adj + ((size_t)rowi * TT + j) * 12;
  float fix = 0.0f;
#pragma unroll
  for (int l = 0; l < 16; ++l) {
    float t = 0.0f;
#pragma unroll
    for (int k = 0; k < 12; ++k) t += av[k] * ap1s[k * 16 + l];
    fix += gelu_f(t) * ap2s[l * 8 + h];
  }
  att += awp[0] * fix;
  atts[j] = att;
  __syncthreads();

  // y = att @ V   (threads 0..63 each own one head-dim column)
  if (j < HD) {
    const float* vb = V + (size_t)(b * TT) * NE + h * HD + j;
    float a2 = 0.0f;
    for (int s2 = 0; s2 < TT; ++s2) a2 += atts[s2] * vb[(size_t)s2 * NE];
    Y[(size_t)rowi * NE + h * HD + j] = a2;
  }
}

// =====================================================================
// Host orchestration
// =====================================================================
extern "C" void kernel_launch(void* const* d_in, const int* in_sizes, int n_in,
                              void* d_out, int out_size, void* d_ws,
                              size_t ws_size, hipStream_t stream) {
  (void)in_sizes; (void)n_in; (void)out_size; (void)ws_size;

  // ---- inputs (setup_inputs dict order, depth-first, insertion order) ----
  const int*   nodes  = (const int*)d_in[0];
  const float* adj    = (const float*)d_in[1];
  const float* nextra = (const float*)d_in[2];
  auto F = [&](int i) { return (const float*)d_in[i]; };
  const float* emb   = F(3);
  const float* fc1w  = F(4),  *fc1b  = F(5);
  const float* bn11g = F(6),  *bn11b = F(7);
  const float* bn12g = F(8),  *bn12b = F(9);
  const float* fc2w  = F(10), *fc2b  = F(11);
  const float* bn21g = F(12), *bn21b = F(13);
  const float* bn22g = F(14), *bn22b = F(15);
  const float* convw = F(16), *convb = F(17);
  const float* posw  = F(18), *posb  = F(19);
  // layers at 20 + 19*l:  ln1.g,ln1.b,ln2.g,ln2.b,q.w,q.b,k.w,k.b,v.w,v.b,
  //                       proj.w,proj.b,ap1,ap2,aw,m1.w,m1.b,m2.w,m2.b
  const float* lnfg = F(96), *lnfb = F(97);
  const float* pew  = F(98), *peb  = F(99);

  // ---- workspace arena (floats) ----
  float* ws = (float*)d_ws;
  size_t off = 0;
  float* valid = ws + off; off += MROWS;
  float* h     = ws + off; off += (size_t)MROWS * 64;
  float* h2    = ws + off; off += (size_t)MROWS * 64;
  float* HI    = ws + off; off += (size_t)MROWS * 128;
  float* HJ    = ws + off; off += (size_t)MROWS * 128;
  float* x     = ws + off; off += (size_t)MROWS * NE;
  float* t0    = ws + off; off += (size_t)MROWS * NE;
  float* big   = ws + off; off += (size_t)MROWS * 2048;  // hdn; aliases Q/K/V/Y
  float* Qb = big;
  float* Kb = big + (size_t)MROWS * NE;
  float* Vb = big + (size_t)MROWS * NE * 2;
  float* Yb = big + (size_t)MROWS * NE * 3;

  auto gemm = [&](const float* A, const float* Bw, const float* bias,
                  const float* resid, const float* rmask, float* C, int M,
                  int N, int K, int lda, int ldb, int ldc, int act) {
    dim3 grid((N + TN - 1) / TN, (M + TM - 1) / TM);
    if ((N % TN) == 0)
      gemm_wmma<false><<<grid, dim3(128), 0, stream>>>(A, Bw, bias, resid,
                                                       rmask, C, M, N, K, lda,
                                                       ldb, ldc, act);
    else
      gemm_wmma<true><<<grid, dim3(128), 0, stream>>>(A, Bw, bias, resid,
                                                      rmask, C, M, N, K, lda,
                                                      ldb, ldc, act);
  };

  // ---- embedding + validity ----
  embed_kernel<<<MROWS, 64, 0, stream>>>(nodes, emb, h, valid);

  // ---- CGCNN layer 1 ----
  gemm(h, fc1w,            fc1b,    nullptr, nullptr, HI, MROWS, 128, 64, 64, 128, 128, 0);
  gemm(h, fc1w + 64 * 128, nullptr, nullptr, nullptr, HJ, MROWS, 128, 64, 64, 128, 128, 0);
  cgcnn_kernel<<<MROWS, 64, 0, stream>>>(h, HI, HJ, adj, valid, fc1w, bn11g,
                                         bn11b, bn12g, bn12b, h2);
  // ---- CGCNN layer 2 ----
  gemm(h2, fc2w,            fc2b,    nullptr, nullptr, HI, MROWS, 128, 64, 64, 128, 128, 0);
  gemm(h2, fc2w + 64 * 128, nullptr, nullptr, nullptr, HJ, MROWS, 128, 64, 64, 128, 128, 0);
  cgcnn_kernel<<<MROWS, 64, 0, stream>>>(h2, HI, HJ, adj, valid, fc2w, bn21g,
                                         bn21b, bn22g, bn22b, h);
  maskh_kernel<<<MROWS, 64, 0, stream>>>(h, valid);

  // ---- conv (64->149, ragged N) + concat extras + pos (512->512) ----
  gemm(h, convw, convb, nullptr, nullptr, t0, MROWS, CONVN, 64, 64, CONVN, NE, 0);
  concat_kernel<<<MROWS, 128, 0, stream>>>(nextra, t0);
  gemm(t0, posw, posb, nullptr, nullptr, x, MROWS, NE, NE, NE, NE, NE, 0);

  // ---- transformer layers ----
  for (int l = 0; l < 4; ++l) {
    int bs = 20 + 19 * l;
    ln_kernel<<<MROWS, 256, 0, stream>>>(x, F(bs + 0), F(bs + 1), t0, NE);
    gemm(t0, F(bs + 4), F(bs + 5), nullptr, nullptr, Qb, MROWS, NE, NE, NE, NE, NE, 0);
    gemm(t0, F(bs + 6), F(bs + 7), nullptr, nullptr, Kb, MROWS, NE, NE, NE, NE, NE, 0);
    gemm(t0, F(bs + 8), F(bs + 9), nullptr, nullptr, Vb, MROWS, NE, NE, NE, NE, NE, 0);
    attn_kernel<<<dim3(TT, NH, BB), dim3(TT), 0, stream>>>(
        Qb, Kb, Vb, valid, adj, F(bs + 12), F(bs + 13), F(bs + 14), Yb);
    gemm(Yb, F(bs + 10), F(bs + 11), x, nullptr, x, MROWS, NE, NE, NE, NE, NE, 0);
    ln_kernel<<<MROWS, 256, 0, stream>>>(x, F(bs + 2), F(bs + 3), t0, NE);
    gemm(t0, F(bs + 15), F(bs + 16), nullptr, nullptr, big, MROWS, 2048, NE, NE, 2048, 2048, 1);
    gemm(big, F(bs + 17), F(bs + 18), x, nullptr, x, MROWS, NE, 2048, 2048, NE, NE, 0);
  }

  // ---- final LN + pe head (512->640), masked by validity ----
  ln_kernel<<<MROWS, 256, 0, stream>>>(x, lnfg, lnfb, t0, NE);
  gemm(t0, pew, peb, nullptr, valid, (float*)d_out, MROWS, 640, NE, NE, 640, 640, 0);
}
